// ModelSRU_8753143349960
// MI455X (gfx1250) — matrix operand order repaired and checked by reading, not compiled, not run on hardware
//
#include <hip/hip_runtime.h>
#include <hip/hip_bf16.h>
#include <stdint.h>

#define Hd    128
#define Bd    256
#define LSEQ  512
#define NL    4
#define MROWS (LSEQ*Bd)   /* 131072 */
#define NT    24          /* 384/16 N tiles */
#define KT    4           /* 128/32 K steps */
#define NGRP  6           /* N-tile groups of 4 (64 cols) per wave */

typedef _Float16 v16h __attribute__((ext_vector_type(16)));
typedef _Float16 v8h  __attribute__((ext_vector_type(8)));
typedef float    v8f  __attribute__((ext_vector_type(8)));

__device__ __forceinline__ float fast_sigmoid(float v) {
  // v_exp_f32 + v_rcp_f32 instead of correctly-rounded divide
  return __builtin_amdgcn_rcpf(1.f + __expf(-v));
}
__device__ __forceinline__ float fast_tanh(float c) {
  // tanh(c) = 1 - 2/(exp(2c)+1); saturates correctly at +-1
  return 1.f - 2.f * __builtin_amdgcn_rcpf(__expf(2.f * c) + 1.f);
}

// ---------------------------------------------------------------------------
// Pack W (layer, K=128, N=384) f32 -> per-lane WMMA B-fragment layout, f16.
// gemm reads: wp[(((layer*NT+nt)*KT+ks)*32+lane)*16 + t]
//   = W[layer][ks*32 + (lane>>4)*16 + t][nt*16 + (lane&15)]
// ---------------------------------------------------------------------------
__global__ __launch_bounds__(256) void sru_pack_w(const float* __restrict__ W,
                                                  _Float16* __restrict__ wp) {
  int gid  = blockIdx.x * 256 + threadIdx.x;   // 4*24*4*32*16 = 196608
  int t    = gid & 15;
  int lane = (gid >> 4) & 31;
  int ks   = (gid >> 9) & 3;
  int rest = gid >> 11;                        // layer*NT + nt
  int nt   = rest % NT;
  int layer= rest / NT;
  int k = ks * 32 + ((lane >> 4) << 4) + t;
  int n = nt * 16 + (lane & 15);
  wp[gid] = (_Float16)W[((size_t)layer * Hd + k) * (3 * Hd) + n];
}

// ---------------------------------------------------------------------------
// Embedding gather: h[l][b][h] = emb[x[b][l]][h]  (f32 + f16 copies)
// ---------------------------------------------------------------------------
__global__ __launch_bounds__(256) void sru_gather(const int* __restrict__ x,
                                                  const float* __restrict__ emb,
                                                  float* __restrict__ h,
                                                  _Float16* __restrict__ h16) {
  size_t gid = (size_t)blockIdx.x * 256 + threadIdx.x;   // MROWS*Hd
  int hh = (int)(gid & (Hd - 1));
  size_t lb = gid >> 7;
  int b = (int)(lb & (Bd - 1));
  int l = (int)(lb >> 8);
  int tok = x[(size_t)b * LSEQ + l];
  float v = emb[(size_t)tok * Hd + hh];
  h[gid]   = v;
  h16[gid] = (_Float16)v;
}

// ---------------------------------------------------------------------------
// U = h16 @ W via v_wmma_f32_16x16x32_f16, fused bias+sigmoid epilogue,
// f16 output (halves the dominant memory stream; U fits in 192MB L2).
// One wave computes a 16x64 tile of U (4 N-tiles): A loaded once per K-step
// and reused 4x -> 16 WMMAs per wave.
// Column group of 64 is region-aligned, so sigmoid branch is wave-uniform.
// ---------------------------------------------------------------------------
__global__ __launch_bounds__(256) void sru_gemm(const _Float16* __restrict__ h16,
                                                const _Float16* __restrict__ wp,
                                                const float* __restrict__ bias,
                                                _Float16* __restrict__ U) {
  int wave = (blockIdx.x << 3) + (threadIdx.x >> 5);   // 8192*6 waves
  int lane = threadIdx.x & 31;
  int mt = wave / NGRP;
  int ng = wave - mt * NGRP;        // 0..5 -> cols [ng*64, ng*64+64)
  int lhalf = lane >> 4;            // 0 or 1
  int lcol  = lane & 15;

  const _Float16* Arow = h16 + ((size_t)mt * 16 + lcol) * Hd + lhalf * 16;
  const _Float16* Bp   = wp + ((size_t)lane) * 16;

  v8f acc0 = {}, acc1 = {}, acc2 = {}, acc3 = {};
#pragma unroll
  for (int ks = 0; ks < KT; ++ks) {
    v8h alo = *(const v8h*)(Arow + ks * 32);
    v8h ahi = *(const v8h*)(Arow + ks * 32 + 8);
    v16h a = __builtin_shufflevector(alo, ahi, 0,1,2,3,4,5,6,7,8,9,10,11,12,13,14,15);
#pragma unroll
    for (int j = 0; j < 4; ++j) {
      const _Float16* bp = Bp + (size_t)(((ng * 4 + j) * KT + ks) * 32) * 16;
      v8h blo = *(const v8h*)(bp);
      v8h bhi = *(const v8h*)(bp + 8);
      v16h bb = __builtin_shufflevector(blo, bhi, 0,1,2,3,4,5,6,7,8,9,10,11,12,13,14,15);
      v8f* accp = (j == 0) ? &acc0 : (j == 1) ? &acc1 : (j == 2) ? &acc2 : &acc3;
      *accp = __builtin_amdgcn_wmma_f32_16x16x32_f16(false, a, false, bb,
                                                     (short)0, *accp, false, false);
    }
  }

  bool dosig = (ng >= 2);           // groups 2..5 are f and r columns
  int row0 = lhalf * 8;
  size_t mbase = (size_t)mt * 16 + row0;
#pragma unroll
  for (int j = 0; j < 4; ++j) {
    const v8f& acc = (j == 0) ? acc0 : (j == 1) ? acc1 : (j == 2) ? acc2 : acc3;
    int col = (ng * 4 + j) * 16 + lcol;
    float bv = dosig ? bias[col - Hd] : 0.f;  // f-bias and r-bias both b[col-128]
#pragma unroll
    for (int r = 0; r < 8; ++r) {
      float v = acc[r] + bv;
      if (dosig) v = fast_sigmoid(v);
      U[(mbase + r) * (3 * Hd) + col] = (_Float16)v;
    }
  }
}

// ---------------------------------------------------------------------------
// SRU scan: one thread per (b, h). c = f*c + (1-f)*xt ; h = r*tanh(c)+(1-r)*h
// U is f16 (sigmoids precomputed); h kept f32 for the residual path.
// In-place on h (each thread owns its element); also writes f16 copy.
// ---------------------------------------------------------------------------
__global__ __launch_bounds__(256) void sru_scan(const _Float16* __restrict__ U,
                                                float* __restrict__ h,
                                                _Float16* __restrict__ h16) {
  int tid = blockIdx.x * 256 + threadIdx.x;   // Bd*Hd = 32768
  int b  = tid >> 7;
  int hh = tid & (Hd - 1);
  float c = 0.f;
  for (int l = 0; l < LSEQ; ++l) {
    size_t rowU = ((size_t)l * Bd + b) * (3 * Hd);
    float xt = (float)U[rowU + hh];
    float f  = (float)U[rowU + Hd + hh];
    float rr = (float)U[rowU + 2 * Hd + hh];
    c = f * c + (1.f - f) * xt;
    size_t ih = ((size_t)l * Bd + b) * Hd + hh;
    float xin = h[ih];
    float ho = rr * fast_tanh(c) + (1.f - rr) * xin;
    h[ih]   = ho;
    h16[ih] = (_Float16)ho;
  }
}

// ---------------------------------------------------------------------------
// out[b][c] = h[L-1][b][:] @ fc_w + fc_b   (256 x 4)
// ---------------------------------------------------------------------------
__global__ __launch_bounds__(256) void sru_fc(const float* __restrict__ h,
                                              const float* __restrict__ fc_w,
                                              const float* __restrict__ fc_b,
                                              float* __restrict__ out) {
  int tid = blockIdx.x * 256 + threadIdx.x;   // 1024
  int b  = tid >> 2;
  int cc = tid & 3;
  const float* row = h + ((size_t)(LSEQ - 1) * Bd + b) * Hd;
  float s = fc_b[cc];
#pragma unroll 8
  for (int k = 0; k < Hd; ++k) s += row[k] * fc_w[k * 4 + cc];
  out[tid] = s;
}

// ---------------------------------------------------------------------------
extern "C" void kernel_launch(void* const* d_in, const int* in_sizes, int n_in,
                              void* d_out, int out_size, void* d_ws, size_t ws_size,
                              hipStream_t stream) {
  const int*   x    = (const int*)d_in[0];
  const float* emb  = (const float*)d_in[1];
  const float* W    = (const float*)d_in[2];
  const float* bvec = (const float*)d_in[3];
  const float* fcw  = (const float*)d_in[4];
  const float* fcb  = (const float*)d_in[5];
  float* out = (float*)d_out;

  char* ws = (char*)d_ws;
  size_t offH   = 0;
  size_t offH16 = offH   + (size_t)MROWS * Hd * sizeof(float);         //  64 MB
  size_t offU   = offH16 + (size_t)MROWS * Hd * sizeof(_Float16);      // +32 MB
  size_t offWP  = offU   + (size_t)MROWS * 3 * Hd * sizeof(_Float16);  // +96 MB
  float*    h   = (float*)   (ws + offH);
  _Float16* h16 = (_Float16*)(ws + offH16);
  _Float16* U   = (_Float16*)(ws + offU);
  _Float16* wp  = (_Float16*)(ws + offWP);                             // +384 KB

  sru_pack_w<<<(NL * NT * KT * 32 * 16) / 256, 256, 0, stream>>>(W, wp);
  sru_gather<<<((size_t)MROWS * Hd) / 256, 256, 0, stream>>>(x, emb, h, h16);

  for (int layer = 0; layer < NL; ++layer) {
    const _Float16* wpl = wp + (size_t)layer * NT * KT * 32 * 16;
    const float*    bl  = bvec + (size_t)layer * 2 * Hd;
    sru_gemm<<<(MROWS / 16) * NGRP / 8, 256, 0, stream>>>(h16, wpl, bl, U);
    sru_scan<<<(Bd * Hd) / 256, 256, 0, stream>>>(U, h, h16);
  }

  sru_fc<<<4, 256, 0, stream>>>(h, fcw, fcb, out);
}
